// ResNetLikeTU_43447889166905
// MI455X (gfx1250) — matrix operand order repaired and checked by reading, not compiled, not run on hardware
//
#include <hip/hip_runtime.h>
#include <hip/hip_bf16.h>

// ---------------------------------------------------------------------------
// ResNet-like sparse-conv block on MI455X (gfx1250, wave32, WMMA).
// f32 -> f16 once; gather via async global->LDS (ASYNCcnt); GEMM with
// v_wmma_f32_16x16x32_f16 (fp32 accumulate); scatter-add with
// global_atomic_add_f32; fused BN+ReLU epilogues.
// ---------------------------------------------------------------------------

typedef _Float16 h16;
typedef __attribute__((ext_vector_type(8)))  _Float16 v8h;
typedef __attribute__((ext_vector_type(16))) _Float16 v16h;
typedef __attribute__((ext_vector_type(8)))  float    v8f;

#define EPSV 1e-5f

#define N_A   50000
#define N_B   200000
#define C_A   256
#define C_B   128
#define C_OUTC 256
#define K_T   8
#define M_T   100000
#define K_C   27
#define M_C   120000

// -------------------------- small utility kernels --------------------------

__global__ void cvt_f32_to_f16_k(const float* __restrict__ src,
                                 h16* __restrict__ dst, int n) {
  int stride = gridDim.x * blockDim.x;
  for (int i = blockIdx.x * blockDim.x + threadIdx.x; i < n; i += stride)
    dst[i] = (h16)src[i];
}

__global__ void zero_f32_k(float* __restrict__ p, int n) {
  int stride = gridDim.x * blockDim.x;
  for (int i = blockIdx.x * blockDim.x + threadIdx.x; i < n; i += stride)
    p[i] = 0.0f;
}

// BN(eval)+ReLU on xa (fp32 accumulator == d_out), write f16 into x_cat[:,0:256],
// then re-zero the accumulator so the second conv can reuse it.
__global__ void bn_relu_concat_zero_k(float* __restrict__ xa,
                                      const float* __restrict__ g,
                                      const float* __restrict__ b,
                                      const float* __restrict__ mu,
                                      const float* __restrict__ var,
                                      h16* __restrict__ xcat) {
  int stride = gridDim.x * blockDim.x;
  for (int i = blockIdx.x * blockDim.x + threadIdx.x; i < N_B * C_A; i += stride) {
    int row = i >> 8;
    int c   = i & 255;
    float s = g[c] * rsqrtf(var[c] + EPSV);
    float y = (xa[i] - mu[c]) * s + b[c];
    y = fmaxf(y, 0.0f);
    xcat[(size_t)row * (C_A + C_B) + c] = (h16)y;
    xa[i] = 0.0f;
  }
}

// feats_b (f32) -> f16 into x_cat[:, 256:384]
__global__ void concat_b_k(const float* __restrict__ fb, h16* __restrict__ xcat) {
  int stride = gridDim.x * blockDim.x;
  for (int i = blockIdx.x * blockDim.x + threadIdx.x; i < N_B * C_B; i += stride) {
    int row = i >> 7;          // /128
    int c   = i & 127;
    xcat[(size_t)row * (C_A + C_B) + C_A + c] = (h16)fb[i];
  }
}

// Final BN(eval)+ReLU in place on d_out (fp32).
__global__ void bn_relu_out_k(float* __restrict__ o,
                              const float* __restrict__ g,
                              const float* __restrict__ b,
                              const float* __restrict__ mu,
                              const float* __restrict__ var) {
  int stride = gridDim.x * blockDim.x;
  for (int i = blockIdx.x * blockDim.x + threadIdx.x; i < N_B * C_OUTC; i += stride) {
    int c = i & 255;
    float s = g[c] * rsqrtf(var[c] + EPSV);
    float y = (o[i] - mu[c]) * s + b[c];
    o[i] = fmaxf(y, 0.0f);
  }
}

// -------------------- CDNA5 async global->LDS gather -----------------------
// 16B per lane, LDS dest addr (low 32 bits of the flat LDS address == LDS
// byte offset), tracked with ASYNCcnt.
__device__ __forceinline__ void async_copy_b128(void* lds_dst, const void* gsrc) {
  unsigned ldsa = (unsigned)(uintptr_t)lds_dst;
  asm volatile("global_load_async_to_lds_b128 %0, %1, off"
               :: "v"(ldsa), "v"(gsrc)
               : "memory");
}
__device__ __forceinline__ void wait_async_zero() {
  asm volatile("s_wait_asynccnt 0x0" ::: "memory");
}

// ------------------- gather -> WMMA GEMM -> scatter-add --------------------
// grid: (tile_split, 4 column quarters of 64, K offsets); block: 128 thr = 4
// waves (wave32). Each wave owns a 16(M) x 16(N) fp32 tile; K swept 32 at a
// time with v_wmma_f32_16x16x32_f16. W slice staged in LDS pre-swizzled into
// B-fragment order so each B fragment is one contiguous 32B read.
template <int CIN>
__global__ void __launch_bounds__(128)
spconv_wmma_k(const h16* __restrict__ feats,      // [n_in, CIN] f16
              const h16* __restrict__ W_all,      // [K, CIN, 256] f16
              const int* __restrict__ in_map,     // [K, M]
              const int* __restrict__ out_map,    // [K, M]
              float* __restrict__ out_acc,        // [n_out, 256] fp32
              int M) {
  constexpr int KSTEPS = CIN / 32;    // WMMA K chunks
  constexpr int CHUNKS = CIN / 8;     // 8-half (16B) chunks per gathered row

  extern __shared__ char smem[];
  h16* lds_w   = (h16*)smem;                                  // CIN*64 halfs
  h16* lds_a   = (h16*)(smem + (size_t)CIN * 64 * 2);         // 16*CIN halfs
  int* lds_in  = (int*)(smem + (size_t)CIN * 64 * 2 + (size_t)CIN * 16 * 2);
  int* lds_out = lds_in + 16;

  const int k    = blockIdx.z;
  const int q    = blockIdx.y;       // output cols [q*64, q*64+64)
  const int tid  = threadIdx.x;
  const int wave = tid >> 5;
  const int lane = tid & 31;
  const int hgrp = lane >> 4;        // lane half-group (ISA fragment layouts)
  const int ln   = lane & 15;

  const h16* Wk = W_all + (size_t)k * CIN * 256;
  const int* im = in_map  + (size_t)k * M;
  const int* om = out_map + (size_t)k * M;

  // Stage W[k][:, q*64 : q*64+64] into LDS in B-fragment order:
  // entry e = (ks*2 + hh)*64 + n  holds  W[ks*32 + hh*16 + j][q*64 + n], j=0..15.
  for (int e = tid; e < KSTEPS * 128; e += 128) {
    int ks = e >> 7;
    int r  = e & 127;
    int hh = r >> 6;
    int n  = r & 63;
    h16* dst = lds_w + (size_t)e * 16;
    const h16* src = Wk + (size_t)(ks * 32 + hh * 16) * 256 + q * 64 + n;
#pragma unroll
    for (int j = 0; j < 16; ++j) dst[j] = src[(size_t)j * 256];
  }
  __syncthreads();

  const int tiles = M >> 4;
  for (int tile = blockIdx.x; tile < tiles; tile += gridDim.x) {
    // Prefetch next tile's map lines while this tile runs.
    int nt = tile + gridDim.x;
    if (nt < tiles && tid < 32) {
      const int* p = (tid < 16) ? (im + nt * 16 + tid) : (om + nt * 16 + (tid - 16));
      __builtin_prefetch(p, 0, 1);
    }
    if (tid < 16)      lds_in[tid]       = __builtin_nontemporal_load(im + tile * 16 + tid);
    else if (tid < 32) lds_out[tid - 16] = __builtin_nontemporal_load(om + tile * 16 + (tid - 16));
    __syncthreads();

    // Async gather of 16 rows (f16) straight into LDS (16B units, ASYNCcnt).
    // 16*CHUNKS is an exact multiple of 128 -> no divergence, EXEC all ones.
#pragma unroll
    for (int it = 0; it < (16 * CHUNKS) / 128; ++it) {
      int c  = it * 128 + tid;
      int r  = c / CHUNKS;
      int jc = c - r * CHUNKS;
      async_copy_b128(lds_a + r * CIN + jc * 8,
                      feats + (size_t)lds_in[r] * CIN + jc * 8);
    }
    wait_async_zero();
    __syncthreads();

    v8f acc = {};
    const h16* arow = lds_a + ln * CIN;   // A row m = lane&15
#pragma unroll
    for (int ks = 0; ks < KSTEPS; ++ks) {
      // A fragment (16x32 f16): lane<16 -> K {base..base+15},
      // lanes>=16 same with base += 16 (hgrp selects).
      const v8h alo = *(const v8h*)(arow + ks * 32 + hgrp * 16);
      const v8h ahi = *(const v8h*)(arow + ks * 32 + hgrp * 16 + 8);
      v16h a = __builtin_shufflevector(alo, ahi,
                 0, 1, 2, 3, 4, 5, 6, 7, 8, 9, 10, 11, 12, 13, 14, 15);
      // B fragment (32x16 f16): col = lane&15, K half selected by hgrp;
      // contiguous thanks to the staged swizzle.
      const h16* bp = lds_w + (size_t)(((ks * 2 + hgrp) * 64) + (wave * 16 + ln)) * 16;
      const v8h blo = *(const v8h*)(bp);
      const v8h bhi = *(const v8h*)(bp + 8);
      v16h b = __builtin_shufflevector(blo, bhi,
                 0, 1, 2, 3, 4, 5, 6, 7, 8, 9, 10, 11, 12, 13, 14, 15);
      acc = __builtin_amdgcn_wmma_f32_16x16x32_f16(
                false, a, false, b, (short)0, acc, false, false);
    }

    // Scatter-add 16x16 fp32 tile: C element (lane, v) = (M = v + 8*hgrp, N = ln).
    const int col = q * 64 + wave * 16 + ln;
#pragma unroll
    for (int v = 0; v < 8; ++v) {
      int m = v + hgrp * 8;
      atomicAdd(out_acc + (size_t)lds_out[m] * 256 + col, acc[v]);
    }
    __syncthreads();
  }
}

// ------------------------------- launcher ----------------------------------

extern "C" void kernel_launch(void* const* d_in, const int* in_sizes, int n_in,
                              void* d_out, int out_size, void* d_ws, size_t ws_size,
                              hipStream_t stream) {
  const float* feats_a    = (const float*)d_in[0];
  const float* feats_b    = (const float*)d_in[1];
  const float* W_t        = (const float*)d_in[2];
  const float* bn_t_gamma = (const float*)d_in[3];
  const float* bn_t_beta  = (const float*)d_in[4];
  const float* bn_t_mean  = (const float*)d_in[5];
  const float* bn_t_var   = (const float*)d_in[6];
  const float* W_c        = (const float*)d_in[7];
  const float* bn_c_gamma = (const float*)d_in[8];
  const float* bn_c_beta  = (const float*)d_in[9];
  const float* bn_c_mean  = (const float*)d_in[10];
  const float* bn_c_var   = (const float*)d_in[11];
  const int*   in_map_t   = (const int*)d_in[12];
  const int*   out_map_t  = (const int*)d_in[13];
  const int*   in_map_c   = (const int*)d_in[14];
  const int*   out_map_c  = (const int*)d_in[15];
  float* out = (float*)d_out;

  // Workspace carve-up (256B aligned).
  char* ws = (char*)d_ws;
  auto carve = [&](size_t bytes) {
    char* p = ws;
    ws += (bytes + 255) & ~(size_t)255;
    return p;
  };
  h16* feats_a16 = (h16*)carve((size_t)N_A * C_A * 2);               // 25.6 MB
  h16* Wt16      = (h16*)carve((size_t)K_T * C_A * C_A * 2);         //  1.0 MB
  h16* Wc16      = (h16*)carve((size_t)K_C * (C_A + C_B) * 256 * 2); //  5.3 MB
  h16* xcat      = (h16*)carve((size_t)N_B * (C_A + C_B) * 2);       // 153.6 MB

  // 1) precision conversion + accumulator init
  cvt_f32_to_f16_k<<<1024, 256, 0, stream>>>(feats_a, feats_a16, N_A * C_A);
  cvt_f32_to_f16_k<<<256, 256, 0, stream>>>(W_t, Wt16, K_T * C_A * C_A);
  cvt_f32_to_f16_k<<<512, 256, 0, stream>>>(W_c, Wc16, K_C * (C_A + C_B) * 256);
  zero_f32_k<<<2048, 256, 0, stream>>>(out, N_B * C_OUTC);

  // 2) transpose conv (k=2^3): cin=256, 8 offsets, 100k pairs each
  {
    dim3 grid(25, 4, K_T);
    size_t lds = (size_t)C_A * 64 * 2 + (size_t)C_A * 16 * 2 + 128;
    spconv_wmma_k<C_A><<<grid, 128, lds, stream>>>(feats_a16, Wt16, in_map_t,
                                                   out_map_t, out, M_T);
  }

  // 3) BN_t + ReLU -> f16 concat; feats_b -> f16 concat; re-zero accumulator
  bn_relu_concat_zero_k<<<2048, 256, 0, stream>>>(out, bn_t_gamma, bn_t_beta,
                                                  bn_t_mean, bn_t_var, xcat);
  concat_b_k<<<1024, 256, 0, stream>>>(feats_b, xcat);

  // 4) 3x3x3 conv: cin=384, 27 offsets, 120k pairs each
  {
    dim3 grid(25, 4, K_C);
    constexpr int cin = C_A + C_B;
    size_t lds = (size_t)cin * 64 * 2 + (size_t)cin * 16 * 2 + 128;
    spconv_wmma_k<cin><<<grid, 128, lds, stream>>>(xcat, Wc16, in_map_c,
                                                   out_map_c, out, M_C);
  }

  // 5) BN_c + ReLU in place
  bn_relu_out_k<<<2048, 256, 0, stream>>>(out, bn_c_gamma, bn_c_beta,
                                          bn_c_mean, bn_c_var);
}
